// CustomLoss_58806692217066
// MI455X (gfx1250) — matrix operand order repaired and checked by reading, not compiled, not run on hardware
//
#include <hip/hip_runtime.h>
#include <hip/hip_bf16.h>

typedef __attribute__((ext_vector_type(16))) _Float16 v16h;
typedef __attribute__((ext_vector_type(8)))  float    v8f;

#define GAMMA_V 0.01f
#define INF_V   1.0e10f
#define SSIM_C1 1.0e-4f
#define SSIM_C2 9.0e-4f
#define HB 118            // blurred spatial size (128 - 11 + 1)
#define PS (118*128)      // stats plane stride (row stride 128)

union AF16 { v16h v; _Float16 h[16]; };
union CF32 { v8f  v; float    f[8];  };

__device__ __forceinline__ void gauss11(float* g) {
    float s = 0.f;
    #pragma unroll
    for (int i = 0; i < 11; ++i) { float d = (i - 5) / 1.5f; g[i] = __expf(-0.5f * d * d); s += g[i]; }
    #pragma unroll
    for (int i = 0; i < 11; ++i) g[i] /= s;
}

// A-matrix (16x32 f16) element->K mapping for lane half (CDNA5 ISA 7.12.2)
__device__ __forceinline__ int a_kidx(int hf, int i) {
    int v = i >> 1;
    return ((v >> 2) << 4) + (hf << 3) + ((v & 3) << 1) + (i & 1);
}

// ---------------- Kernel 1: per-image blur stats: mu = blur(x), m2 = blur(x^2)
__global__ __launch_bounds__(256) void stats_kernel(const float* __restrict__ img,
                                                    const float* __restrict__ simg,
                                                    float* __restrict__ mu,
                                                    float* __restrict__ m2) {
    __shared__ float plane[128 * 128];            // 64 KB
    const int planeId = blockIdx.x;               // 0..119
    const int imgIdx = planeId / 3, c = planeId % 3;
    const float* src = (imgIdx < 32) ? img  + (imgIdx * 3 + c) * 16384
                                     : simg + ((imgIdx - 32) * 3 + c) * 16384;
    for (int i = threadIdx.x; i < 16384; i += 256) plane[i] = src[i];
    __syncthreads();
    float g[11]; gauss11(g);
    for (int o = threadIdx.x; o < HB * HB; o += 256) {
        const int r = o / HB, w = o % HB;
        float a0 = 0.f, a1 = 0.f;
        #pragma unroll
        for (int i = 0; i < 11; ++i) {
            float h0 = 0.f, h1 = 0.f;
            const float* row = &plane[(r + i) * 128 + w];
            #pragma unroll
            for (int j = 0; j < 11; ++j) { float v = row[j]; h0 += g[j] * v; h1 += g[j] * v * v; }
            a0 += g[i] * h0; a1 += g[i] * h1;
        }
        mu[planeId * PS + r * 128 + w] = a0;
        m2[planeId * PS + r * 128 + w] = a1;
    }
}

// ---------------- Kernel 2: per-pair blur(x*y) via WMMA band-matrix GEMMs + SSIM reduce
__global__ __launch_bounds__(256) void pair_ssim_kernel(const float* __restrict__ img,
                                                        const float* __restrict__ simg,
                                                        const float* __restrict__ mu,
                                                        const float* __restrict__ m2,
                                                        float* __restrict__ pairPartial) {
    __shared__ _Float16 prodLDS[128 * 128];       // 32 KB
    __shared__ _Float16 midLDS [128 * 128];       // 32 KB
    const int id = blockIdx.x;                    // 0..767  (pair*3 + channel)
    const int p = id / 3, c = id % 3;
    const int b = p >> 3, s = p & 7;
    const float* xs = simg + (s * 3 + c) * 16384;
    const float* yb = img  + (b * 3 + c) * 16384;
    for (int i = threadIdx.x; i < 16384; i += 256)
        prodLDS[i] = (_Float16)(xs[i] * yb[i]);
    __syncthreads();

    float g[11]; gauss11(g);
    const int lane = threadIdx.x & 31;
    const int wave = threadIdx.x >> 5;            // 8 waves per block
    const int hf = lane >> 4, lo = lane & 15;

    // horizontal band B: B[k][n] = g[k-n];  lane: N=lo, K per a_kidx
    AF16 bh;
    #pragma unroll
    for (int i = 0; i < 16; ++i) {
        int K = a_kidx(hf, i), t = K - lo;
        bh.h[i] = (_Float16)((t >= 0 && t <= 10) ? g[t] : 0.f);
    }
    // vertical band A: A[m][k] = g[k-m];  lane: M=lo, K per a_kidx
    AF16 av;
    #pragma unroll
    for (int i = 0; i < 16; ++i) {
        int t = a_kidx(hf, i) - lo;
        av.h[i] = (_Float16)((t >= 0 && t <= 10) ? g[t] : 0.f);
    }

    // Stage 1: horizontal blur (each wave owns a 16-row band)
    const int r0 = wave * 16;
    for (int ct = 0; ct < 8; ++ct) {
        const int w0 = ct * 16;
        AF16 a;
        #pragma unroll
        for (int i = 0; i < 16; ++i) {
            int col = w0 + a_kidx(hf, i);
            a.h[i] = (col < 128) ? prodLDS[(r0 + lo) * 128 + col] : (_Float16)0.f;
        }
        CF32 d; d.v = (v8f){};
        d.v = __builtin_amdgcn_wmma_f32_16x16x32_f16(false, a.v, false, bh.v,
                                                     (short)0, d.v, false, false);
        #pragma unroll
        for (int r = 0; r < 8; ++r)
            midLDS[(r0 + hf * 8 + r) * 128 + (w0 + lo)] = (_Float16)d.f[r];
    }
    __syncthreads();

    // Stage 2: vertical blur + per-pixel SSIM
    const float* mux = mu + ((32 + s) * 3 + c) * PS;
    const float* m2x = m2 + ((32 + s) * 3 + c) * PS;
    const float* muy = mu + (b * 3 + c) * PS;
    const float* m2y = m2 + (b * 3 + c) * PS;
    float acc = 0.f;
    const int ob = wave;                          // output-row band
    for (int ct = 0; ct < 8; ++ct) {
        const int w0 = ct * 16;
        AF16 bv;
        #pragma unroll
        for (int i = 0; i < 16; ++i) {
            int row = ob * 16 + a_kidx(hf, i); if (row > 127) row = 127;
            bv.h[i] = midLDS[row * 128 + (w0 + lo)];
        }
        CF32 d; d.v = (v8f){};
        d.v = __builtin_amdgcn_wmma_f32_16x16x32_f16(false, av.v, false, bv.v,
                                                     (short)0, d.v, false, false);
        #pragma unroll
        for (int r = 0; r < 8; ++r) {
            const int row = ob * 16 + hf * 8 + r, col = w0 + lo;
            if (row < HB && col < HB) {
                const int o = row * 128 + col;
                float mx = mux[o], my = muy[o];
                float sxx = m2x[o] - mx * mx;
                float syy = m2y[o] - my * my;
                float sxy = d.f[r] - mx * my;
                float num = (2.f * mx * my + SSIM_C1) * (2.f * sxy + SSIM_C2);
                float den = (mx * mx + my * my + SSIM_C1) * (sxx + syy + SSIM_C2);
                acc += num / den;
            }
        }
    }
    __syncthreads();
    float* red = (float*)prodLDS;                 // reuse LDS for reduction
    red[threadIdx.x] = acc;
    __syncthreads();
    for (int off = 128; off > 0; off >>= 1) {
        if (threadIdx.x < off) red[threadIdx.x] += red[threadIdx.x + off];
        __syncthreads();
    }
    if (threadIdx.x == 0) pairPartial[id] = red[0];
}

// ---------------- Kernel 3: soft-DTW wavefront, one block per (b,s) pair
__global__ __launch_bounds__(256) void sdtw_kernel(const float* __restrict__ x,
                                                   const float* __restrict__ y,
                                                   float* __restrict__ dtwOut) {
    __shared__ float buf[3][257];                 // rolling anti-diagonals, slot 0 == INF
    __shared__ float ys[256];
    const int p = blockIdx.x;                     // 0..255
    const int b = p >> 3, s = p & 7;
    const int i = threadIdx.x;                    // row index owned by this thread
    ys[i] = y[s * 256 + i];
    const float xi = x[b * 256 + i];
    buf[0][i] = INF_V; buf[1][i] = INF_V; buf[2][i] = INF_V;
    if (i == 0) { buf[0][256] = INF_V; buf[1][256] = INF_V; buf[2][256] = INF_V; }
    __syncthreads();
    const float invg = 1.f / GAMMA_V;
    float r = INF_V;
    for (int k = 0; k < 511; ++k) {
        float* cur = buf[k % 3];
        float* p1  = buf[(k + 2) % 3];
        float* p2  = buf[(k + 1) % 3];
        float a  = p2[i];       // R[i-1][j-1]
        float bb = p1[i];       // R[i-1][j]
        float cc = p1[i + 1];   // R[i][j-1]
        float m = fminf(a, fminf(bb, cc));
        float sm = m - GAMMA_V * __logf(__expf(-(a - m) * invg) +
                                        __expf(-(bb - m) * invg) +
                                        __expf(-(cc - m) * invg));
        if (k == 0 && i == 0) sm = 0.f;
        const int j = k - i;
        if (j >= 0 && j < 256) {
            float diff = xi - ys[j];
            r = diff * diff + sm;
        } else r = INF_V;
        cur[i + 1] = r;
        __syncthreads();
    }
    if (i == 255) dtwOut[p] = r;                  // R[T-1][T-1]
}

// ---------------- Kernel 4: BCE + combine into (loss, bce, metric_sum)
__global__ __launch_bounds__(256) void finalize_kernel(const float* __restrict__ out,
                                                       const float* __restrict__ lab,
                                                       const float* __restrict__ pairPartial,
                                                       const float* __restrict__ dtw,
                                                       float* __restrict__ result) {
    __shared__ float redM[32];
    __shared__ float redB[32];
    const int t = threadIdx.x;
    if (t < 32) {
        float accm = 0.f;
        for (int s = 0; s < 8; ++s) {
            int p = t * 8 + s;
            float ssum = pairPartial[p * 3 + 0] + pairPartial[p * 3 + 1] + pairPartial[p * 3 + 2];
            float ssim = ssum / (3.0f * HB * HB);
            float m0 = (1.f - ssim) * 10.f;
            float m1 = dtw[p] / 500.f;
            accm += 0.5f * (m0 + m1);
        }
        redM[t] = accm / 8.f;                     // per_b
        float o = out[t], l = lab[t];
        float ln  = fmaxf(__logf(o), -100.f);
        float ln1 = fmaxf(log1pf(-o), -100.f);
        redB[t] = -(l * ln + (1.f - l) * ln1);
    }
    __syncthreads();
    if (t == 0) {
        float ms = 0.f, bs = 0.f;
        for (int i = 0; i < 32; ++i) { ms += redM[i]; bs += redB[i]; }
        float bce = bs / 32.f;
        result[0] = bce + ms / 10.f;              // loss
        result[1] = bce;
        result[2] = ms;                           // metric_sum
    }
}

extern "C" void kernel_launch(void* const* d_in, const int* in_sizes, int n_in,
                              void* d_out, int out_size, void* d_ws, size_t ws_size,
                              hipStream_t stream) {
    const float* input   = (const float*)d_in[0];  // [32,256]
    const float* samples = (const float*)d_in[1];  // [8,256]
    const float* img     = (const float*)d_in[2];  // [32,3,128,128]
    const float* simg    = (const float*)d_in[3];  // [8,3,128,128]
    const float* output  = (const float*)d_in[4];  // [32,1]
    const float* labels  = (const float*)d_in[5];  // [32,1]

    float* ws = (float*)d_ws;
    float* mu = ws;                        // 120 * PS
    float* m2 = mu + 120 * PS;             // 120 * PS
    float* pairPartial = m2 + 120 * PS;    // 768
    float* dtw = pairPartial + 768;        // 256

    stats_kernel<<<120, 256, 0, stream>>>(img, simg, mu, m2);
    pair_ssim_kernel<<<768, 256, 0, stream>>>(img, simg, mu, m2, pairPartial);
    sdtw_kernel<<<256, 256, 0, stream>>>(input, samples, dtw);
    finalize_kernel<<<1, 256, 0, stream>>>(output, labels, pairPartial, dtw, (float*)d_out);
}
